// FullAttention_20057497272723
// MI455X (gfx1250) — matrix operand order repaired and checked by reading, not compiled, not run on hardware
//
#include <hip/hip_runtime.h>
#include <math.h>

#define B_ 4
#define L_ 2048
#define S_ 2048
#define H_ 16
#define E_ 64
#define D_ 64

typedef __attribute__((ext_vector_type(16))) _Float16 v16h;
typedef __attribute__((ext_vector_type(8)))  float    v8f;

// LDS row stride in f16 elements: 64 data + 8 pad = 72 (144 B, 16B aligned)
#define TS 72
#define LOG2E 1.44269504088896340736f

static __device__ __forceinline__ v16h ld_frag(const _Float16* p0, const _Float16* p1) {
    union { v16h v; uint4 q[2]; } u;
    u.q[0] = *(const uint4*)p0;   // 8 halfs (chunk 0)
    u.q[1] = *(const uint4*)p1;   // 8 halfs (chunk 1)
    return u.v;
}

static __device__ __forceinline__ void st4_h(_Float16* p, float4 v) {
    union { _Float16 h[4]; uint2 u; } t;
    t.h[0] = (_Float16)v.x; t.h[1] = (_Float16)v.y;
    t.h[2] = (_Float16)v.z; t.h[3] = (_Float16)v.w;
    *(uint2*)p = t.u;
}

__global__ __launch_bounds__(256) void fattn_wmma_kernel(
    const float* __restrict__ Q, const float* __restrict__ K,
    const float* __restrict__ V, float* __restrict__ O)
{
    const int lt   = blockIdx.x;       // 128-row query tile
    const int h    = blockIdx.y;
    const int b    = blockIdx.z;
    const int tid  = threadIdx.x;
    const int lane = tid & 31;
    const int wave = tid >> 5;         // 8 waves, 16 query rows each
    const int hl   = lane & 15;        // lane within 16-half
    const int hs   = lane >> 4;        // which 16-lane half

    __shared__ _Float16 Qs [128 * TS];     // [row][e]   (pre-scaled by log2e)
    __shared__ _Float16 Ks [64 * TS];      // [key][e]
    __shared__ _Float16 Vts[64 * TS];      // [d][key]  (transposed)
    __shared__ _Float16 Ps [8 * 16 * TS];  // per-wave P scratch [row][key]

    const size_t rowstride = (size_t)H_ * E_;  // floats between seq positions
    const size_t qbase = (((size_t)b * L_ + (size_t)lt * 128) * H_ + h) * (size_t)E_;
    const size_t kbase = (((size_t)b * S_) * H_ + h) * (size_t)E_;

    // per-thread cooperative-load coordinates: fixed column, rows r0 + 16*i
    const int r0 = tid >> 4;               // 0..15
    const int c4 = (tid & 15) * 4;         // 0..60

    // ---- load Q tile: 128x64 fp32 -> f16 LDS, row-major, x log2(e) ----
#pragma unroll
    for (int i = 0; i < 8; ++i) {
        int row = r0 + i * 16;             // 0..127
        float4 v = *(const float4*)(Q + qbase + (size_t)row * rowstride + c4);
        v.x *= LOG2E; v.y *= LOG2E; v.z *= LOG2E; v.w *= LOG2E;
        st4_h(&Qs[row * TS + c4], v);
    }
    __syncthreads();

    // ---- Q A-fragments for this wave's 16 rows (E split into two K=32 windows) ----
    v16h qf[2];
    {
        const _Float16* qrow = &Qs[(wave * 16 + hl) * TS];
#pragma unroll
        for (int eb = 0; eb < 2; ++eb) {
            int e0 = eb * 32 + 16 * hs;      // A layout: 16 consecutive e at +0 and +8
            qf[eb] = ld_frag(qrow + e0, qrow + e0 + 8);
        }
    }

    v8f acc[4];
#pragma unroll
    for (int t = 0; t < 4; ++t) acc[t] = v8f{};
    float m_i[8], l_i[8];                    // running max (log2 domain), per-LANE partial sum
#pragma unroll
    for (int r = 0; r < 8; ++r) { m_i[r] = -INFINITY; l_i[r] = 0.0f; }

    _Float16* pw = &Ps[wave * 16 * TS];

    // ---- prologue: stage tile 0 (K and V) into registers ----
    const float* pKt = K + kbase + (size_t)r0 * rowstride + c4;
    const float* pVt = V + kbase + (size_t)r0 * rowstride + c4;
    float4 kreg[4], vreg[4];
#pragma unroll
    for (int i = 0; i < 4; ++i) {
        kreg[i] = *(const float4*)(pKt + (size_t)(i * 16) * rowstride);
        vreg[i] = *(const float4*)(pVt + (size_t)(i * 16) * rowstride);
    }

    for (int s0 = 0; s0 < S_; s0 += 64) {
        __syncthreads();                     // previous iteration's LDS reads done
        // ---- commit staged tile to LDS (K row-major, V transposed), fp32 -> f16 ----
#pragma unroll
        for (int i = 0; i < 4; ++i) {
            int row = r0 + i * 16;           // key within tile
            st4_h(&Ks[row * TS + c4], kreg[i]);
            Vts[(c4 + 0) * TS + row] = (_Float16)vreg[i].x;
            Vts[(c4 + 1) * TS + row] = (_Float16)vreg[i].y;
            Vts[(c4 + 2) * TS + row] = (_Float16)vreg[i].z;
            Vts[(c4 + 3) * TS + row] = (_Float16)vreg[i].w;
        }
        // ---- prefetch next tile into registers (latency hides behind compute) ----
        if (s0 + 64 < S_) {
            const float* pKn = pKt + (size_t)(s0 + 64) * rowstride;
            const float* pVn = pVt + (size_t)(s0 + 64) * rowstride;
#pragma unroll
            for (int i = 0; i < 4; ++i) {
                kreg[i] = *(const float4*)(pKn + (size_t)(i * 16) * rowstride);
                vreg[i] = *(const float4*)(pVn + (size_t)(i * 16) * rowstride);
            }
        }
        __syncthreads();

        // ---- scores S' = (Q*log2e) * K^T  (4 key tiles x 2 K-steps = 8 WMMA) ----
        v8f sc[4];
#pragma unroll
        for (int t = 0; t < 4; ++t) {
            sc[t] = v8f{};
#pragma unroll
            for (int eb = 0; eb < 2; ++eb) {
                // B layout: col N = key = t*16+hl; 16 consecutive e at eb*32 + 16*hs
                const _Float16* kp = &Ks[(t * 16 + hl) * TS + eb * 32 + 16 * hs];
                v16h bf = ld_frag(kp, kp + 8);
                sc[t] = __builtin_amdgcn_wmma_f32_16x16x32_f16(
                    false, qf[eb], false, bf, (short)0, sc[t], false, false);
            }
        }

        // ---- online softmax (base-2): row = r + 8*hs ----
        // row max must be row-uniform -> 16-lane butterfly
        float bm[8];
#pragma unroll
        for (int r = 0; r < 8; ++r) {
            float v = fmaxf(fmaxf(sc[0][r], sc[1][r]), fmaxf(sc[2][r], sc[3][r]));
#pragma unroll
            for (int off = 1; off <= 8; off <<= 1)
                v = fmaxf(v, __shfl_xor(v, off, 32));
            bm[r] = v;
        }
#pragma unroll
        for (int r = 0; r < 8; ++r) {
            float mn    = fmaxf(m_i[r], bm[r]);
            float scale = exp2f(m_i[r] - mn);    // bare v_exp_f32
            m_i[r] = mn;
            l_i[r] *= scale;                     // per-lane partial sum rescale
#pragma unroll
            for (int t = 0; t < 4; ++t) acc[t][r] *= scale;
        }
        // P = exp2(S' - m); accumulate per-LANE partial row sums (reduced at epilogue)
#pragma unroll
        for (int t = 0; t < 4; ++t) {
#pragma unroll
            for (int r = 0; r < 8; ++r) {
                float p = exp2f(sc[t][r] - m_i[r]);
                l_i[r] += p;
                pw[(r + 8 * hs) * TS + t * 16 + hl] = (_Float16)p;
            }
        }

        // ---- O += P * V  (2 K-steps x 4 d tiles = 8 WMMA) ----
#pragma unroll
        for (int kb = 0; kb < 2; ++kb) {
            // A layout: row = hl; 16 consecutive keys at kb*32 + 16*hs
            const _Float16* ap = &pw[hl * TS + kb * 32 + 16 * hs];
            v16h af = ld_frag(ap, ap + 8);
#pragma unroll
            for (int t = 0; t < 4; ++t) {
                // B layout: col N = d = t*16+hl; 16 consecutive keys at kb*32 + 16*hs
                const _Float16* vp = &Vts[(t * 16 + hl) * TS + kb * 32 + 16 * hs];
                v16h bf = ld_frag(vp, vp + 8);
                acc[t] = __builtin_amdgcn_wmma_f32_16x16x32_f16(
                    false, af, false, bf, (short)0, acc[t], false, false);
            }
        }
    }

    // ---- epilogue: single cross-lane reduction of l, normalize, store ----
    float inv[8];
#pragma unroll
    for (int r = 0; r < 8; ++r) {
        float v = l_i[r];
#pragma unroll
        for (int off = 1; off <= 8; off <<= 1)
            v += __shfl_xor(v, off, 32);
        inv[r] = 1.0f / v;
    }
    const int l0 = lt * 128 + wave * 16;
#pragma unroll
    for (int t = 0; t < 4; ++t) {
#pragma unroll
        for (int r = 0; r < 8; ++r) {
            int lrow = l0 + r + 8 * hs;
            int d    = t * 16 + hl;
            O[(((size_t)b * L_ + lrow) * H_ + h) * (size_t)D_ + d] = acc[t][r] * inv[r];
        }
    }
}

extern "C" void kernel_launch(void* const* d_in, const int* in_sizes, int n_in,
                              void* d_out, int out_size, void* d_ws, size_t ws_size,
                              hipStream_t stream) {
    (void)in_sizes; (void)n_in; (void)out_size; (void)d_ws; (void)ws_size;
    const float* Q = (const float*)d_in[0];
    const float* K = (const float*)d_in[1];
    const float* V = (const float*)d_in[2];
    float* O = (float*)d_out;
    dim3 grid(L_ / 128, H_, B_);  // 16 x 16 x 4 = 1024 workgroups
    dim3 block(256, 1, 1);        // 8 waves (wave32), 16 query rows each
    fattn_wmma_kernel<<<grid, block, 0, stream>>>(Q, K, V, O);
}
